// FLAPaTH_89498528514099
// MI455X (gfx1250) — matrix operand order repaired and checked by reading, not compiled, not run on hardware
//
#include <hip/hip_runtime.h>
#include <hip/hip_bf16.h>

typedef __attribute__((ext_vector_type(16))) _Float16 v16h;
typedef __attribute__((ext_vector_type(8)))  float    v8f;
typedef int v4i_vs __attribute__((vector_size(4 * sizeof(int))));   // matches builtin param

#define DEV __device__ __forceinline__

constexpr int Bc = 2, Tc = 1024, Dc = 1024, Hc = 16, HDc = 64, RKc = 32;
constexpr int BTc = Bc * Tc;   // 2048
constexpr int BHc = Bc * Hc;   // 32

union FragH { v16h v; unsigned u[8]; _Float16 h[16]; };

DEV v8f zf() { v8f z = {0.f,0.f,0.f,0.f,0.f,0.f,0.f,0.f}; return z; }

// A-frag (16x32, f16): lane m = lane&15, half = lane>>4.
// VGPR v holds K pair: K = ((v>>2)<<4) + 8*half + ((v&3)<<1) + {0,1}
DEV int a_koff(int v, int half) { return ((v >> 2) << 4) + 8 * half + ((v & 3) << 1); }
// B-frag (32x16, f16): lane n = lane&15. VGPR v holds K = ((v>>2)<<4) + 8*half + ((v&3)<<1) + {0,1}
DEV int b_koff(int v, int half) { return ((v >> 2) << 4) + 8 * half + ((v & 3) << 1); }

DEV unsigned ld2h(const _Float16* p) { return *reinterpret_cast<const unsigned*>(p); }

DEV v8f wmma_f16(const FragH& a, const FragH& b, v8f c) {
    return __builtin_amdgcn_wmma_f32_16x16x32_f16(false, a.v, false, b.v,
                                                  (short)0, c, false, false);
}

// ---- CDNA5 async global->LDS path (guarded; falls back to plain LDS copy) ----
#if defined(__has_builtin)
#  if __has_builtin(__builtin_amdgcn_global_load_async_to_lds_b128) && \
      __has_builtin(__builtin_amdgcn_s_wait_asynccnt)
#    define USE_ASYNC_LDS 1
#  endif
#endif
#ifndef USE_ASYNC_LDS
#  define USE_ASYNC_LDS 0
#endif

// Copy 16 bytes global -> LDS (one async b128 per thread).
DEV void cp16_g2s(const _Float16* __restrict__ src, _Float16* dst) {
#if USE_ASYNC_LDS
    __builtin_amdgcn_global_load_async_to_lds_b128(
        (__attribute__((address_space(1))) v4i_vs*)(src),
        (__attribute__((address_space(3))) v4i_vs*)(dst), 0, 0);
#else
    *reinterpret_cast<uint4*>(dst) = *reinterpret_cast<const uint4*>(src);
#endif
}
DEV void async_join() {
#if USE_ASYNC_LDS
    __builtin_amdgcn_s_wait_asynccnt(0);
#endif
}

// ---------------------------------------------------------------- casts
__global__ void cast_f16_k(const float* __restrict__ s, _Float16* __restrict__ d, int n) {
    for (int i = blockIdx.x * blockDim.x + threadIdx.x; i < n; i += gridDim.x * blockDim.x)
        d[i] = (_Float16)s[i];
}
// src [R][C] f32 -> dst [C][R] f16 (column-major weight for contiguous B-frags)
__global__ void cast_tr_k(const float* __restrict__ s, _Float16* __restrict__ d, int R, int C) {
    for (int i = blockIdx.x * blockDim.x + threadIdx.x; i < R * C; i += gridDim.x * blockDim.x) {
        int r = i / C, c = i % C;
        d[(size_t)c * R + r] = (_Float16)s[i];
    }
}

// ---------------------------------------------------------------- big GEMM (f16 WMMA, f32 accum)
// Wave computes 16x64 (4 tiles, shared A-frag); WG = 8 waves stacked in M -> 128x64 tile.
// B tile (32K x 64N) staged in LDS, double-buffered via async global->LDS copies.
// AMODE: 0 = A row-major [M][K] (lda); 1 = A is O in [B,H,T,HD] with m=b*T+t, k=h*64+d
// OMODE: 0 = f16 [M][N]; 1 = f16 [B,H,T,HD]; 2 = f16 [B,H,HD,T]; 3 = f32 [M][N]
// Requires M%128==0, N%64==0, K%32==0.
template <int AMODE, int OMODE>
__global__ __launch_bounds__(256) void gemm_wmma2_k(
    const _Float16* __restrict__ A, const _Float16* __restrict__ Bt,
    _Float16* __restrict__ Oh, float* __restrict__ Of,
    int M, int N, int K, int lda, int ldb)
{
    __shared__ _Float16 Bs[2][64 * 32];   // [buf][n_loc][k_loc]
    const int tid = threadIdx.x;
    const int lane = tid & 31, wv = tid >> 5;
    const int half = lane >> 4, l16 = lane & 15;
    const int M0 = blockIdx.x * 128 + wv * 16;
    const int N0 = blockIdx.y * 64;
    const int m = M0 + l16;
    const int sn = tid >> 2, sk = (tid & 3) * 8;   // staging: 16B per thread
    int bb = 0, tt = 0;
    if (AMODE == 1) { bb = m >> 10; tt = m & 1023; }

    v8f acc[4];
#pragma unroll
    for (int nt2 = 0; nt2 < 4; ++nt2) acc[nt2] = zf();

    cp16_g2s(Bt + (size_t)(N0 + sn) * ldb + sk, &Bs[0][sn * 32 + sk]);
    async_join();
    __syncthreads();

    int buf = 0;
    for (int k0 = 0; k0 < K; k0 += 32) {
        if (k0 + 32 < K)
            cp16_g2s(Bt + (size_t)(N0 + sn) * ldb + k0 + 32 + sk,
                     &Bs[buf ^ 1][sn * 32 + sk]);
        if (AMODE == 0 && k0 + 32 < K)
            __builtin_prefetch(A + (size_t)m * lda + k0 + 32, 0, 1);
        FragH a;
#pragma unroll
        for (int v = 0; v < 8; ++v) {
            int ka = k0 + a_koff(v, half);
            if (AMODE == 0) {
                a.u[v] = ld2h(A + (size_t)m * lda + ka);
            } else {
                int hh = ka >> 6, dd = ka & 63;
                a.u[v] = ld2h(A + (((size_t)(bb * Hc + hh) * Tc + tt) << 6) + dd);
            }
        }
#pragma unroll
        for (int nt2 = 0; nt2 < 4; ++nt2) {
            FragH b;
#pragma unroll
            for (int v = 0; v < 8; ++v)
                b.u[v] = *reinterpret_cast<const unsigned*>(
                    &Bs[buf][(nt2 * 16 + l16) * 32 + b_koff(v, half)]);
            acc[nt2] = wmma_f16(a, b, acc[nt2]);
        }
        async_join();
        __syncthreads();
        buf ^= 1;
    }
#pragma unroll
    for (int nt2 = 0; nt2 < 4; ++nt2) {
        const int n = N0 + nt2 * 16 + l16;
#pragma unroll
        for (int g = 0; g < 8; ++g) {
            int row = M0 + g + 8 * half;
            float val = acc[nt2][g];
            if (OMODE == 0) {
                Oh[(size_t)row * N + n] = (_Float16)val;
            } else if (OMODE == 3) {
                Of[(size_t)row * N + n] = val;
            } else {
                int b2 = row >> 10, t2 = row & 1023, hh = n >> 6, dd = n & 63;
                if (OMODE == 1)
                    Oh[(((size_t)(b2 * Hc + hh) * Tc + t2) << 6) + dd] = (_Float16)val;
                else
                    Oh[(((size_t)(b2 * Hc + hh) * HDc + dd) << 10) + t2] = (_Float16)val;
            }
        }
    }
}

// ---------------------------------------------------------------- small-N GEMM (no LDS)
// OMODE 0 only; used for N=32 / N=16 projections.
__global__ __launch_bounds__(256) void gemm_small_k(
    const _Float16* __restrict__ A, const _Float16* __restrict__ Bt,
    _Float16* __restrict__ Oh, int M, int N, int K, int lda, int ldb)
{
    const int lane = threadIdx.x & 31, wv = threadIdx.x >> 5;
    const int half = lane >> 4, l16 = lane & 15;
    const int M0 = (blockIdx.x * 2 + (wv & 1)) * 16;
    const int N0 = (blockIdx.y * 4 + (wv >> 1)) * 16;
    if (N0 >= N || M0 >= M) return;
    const int m = M0 + l16, n = N0 + l16;
    v8f acc = zf();
    for (int k0 = 0; k0 < K; k0 += 32) {
        if (k0 + 32 < K) __builtin_prefetch(A + (size_t)m * lda + k0 + 32, 0, 1);
        FragH a, b;
#pragma unroll
        for (int v = 0; v < 8; ++v) {
            a.u[v] = ld2h(A + (size_t)m * lda + k0 + a_koff(v, half));
            b.u[v] = ld2h(Bt + (size_t)n * ldb + k0 + b_koff(v, half));
        }
        acc = wmma_f16(a, b, acc);
    }
#pragma unroll
    for (int g = 0; g < 8; ++g)
        Oh[(size_t)(M0 + g + 8 * half) * N + n] = (_Float16)acc[g];
}

// ---------------------------------------------------------------- conv + silu + unit-norm
// w0 [B,T,D] f16 -> wp [B,H,T,HD] f16. One wave per (b,t,h), 2 channels per lane.
__global__ __launch_bounds__(256) void conv_silu_norm_k(
    const _Float16* __restrict__ w0, const float* __restrict__ cw, _Float16* __restrict__ wp)
{
    int gid = blockIdx.x * blockDim.x + threadIdx.x;
    int wave = gid >> 5, lane = gid & 31;
    int h = wave % Hc, bt = wave / Hc, t = bt % Tc, b = bt / Tc;
    float val[2]; float ss = 0.f;
#pragma unroll
    for (int j = 0; j < 2; ++j) {
        int dl = lane * 2 + j, dg = h * HDc + dl;
        const _Float16* base = w0 + ((size_t)(b * Tc + t)) * Dc + dg;
        const float* c3 = cw + dg * 3;
        float s = (float)base[0] * c3[2];
        if (t >= 1) s += (float)base[-Dc] * c3[1];
        if (t >= 2) s += (float)base[-2 * Dc] * c3[0];
        s = s / (1.f + __expf(-s));   // silu
        val[j] = s; ss += s * s;
    }
    for (int m2 = 1; m2 < 32; m2 <<= 1) ss += __shfl_xor(ss, m2);
    float inv = 1.f / fmaxf(sqrtf(ss), 1e-12f);
    size_t ob = (((size_t)(b * Hc + h) * Tc + t) << 6) + lane * 2;
    wp[ob]     = (_Float16)(val[0] * inv);
    wp[ob + 1] = (_Float16)(val[1] * inv);
}

// ---------------------------------------------------------------- beta = 2*sigmoid(logits)
__global__ void beta_k(const _Float16* __restrict__ bl, float* __restrict__ bf) {
    int i = blockIdx.x * blockDim.x + threadIdx.x;
    if (i >= BTc * Hc) return;
    int h = i & 15, m = i >> 4, b = m >> 10, t = m & 1023;
    bf[((b * Hc + h) << 10) + t] = 2.f / (1.f + __expf(-(float)bl[i]));
}

// ---------------------------------------------------------------- blocked forward substitution
// Solves C'[i,j] = beta_i*(Ms[i,j] - sum_{t<i} M[i,t]*C'[t,j]) per (b,h).
// C' stored TRANSPOSED: Ct[j*T + i], f16. One WG (16 waves) per head; block rows of 16.
__global__ __launch_bounds__(512) void solve_k(
    const _Float16* __restrict__ wp, const float* __restrict__ betaf, _Float16* __restrict__ Cws)
{
    __shared__ _Float16 Ms[64 * 256];   // M[I, kb] tiles: Ms[kb*256 + i*16 + t]
    __shared__ float betas[Tc];
    const int bh = blockIdx.x, tid = threadIdx.x;
    const int lane = tid & 31, wv = tid >> 5, half = lane >> 4, l16 = lane & 15;
    const _Float16* W = wp + (size_t)bh * Tc * HDc;
    _Float16* Ct = Cws + (size_t)bh * Tc * Tc;
    for (int i = tid; i < Tc; i += 512) betas[i] = betaf[bh * Tc + i];
    __syncthreads();

    for (int I = 0; I < 64; ++I) {
        // phase 1: M strip  M[I,kb] = W_I @ W_kb^T  (K=64 -> 2 WMMAs)
        for (int kb = wv; kb <= I; kb += 16) {
            v8f acc = zf();
#pragma unroll
            for (int c = 0; c < 2; ++c) {
                FragH a, b;
#pragma unroll
                for (int v = 0; v < 8; ++v) {
                    a.u[v] = ld2h(W + (size_t)(I * 16 + l16) * HDc + 32 * c + a_koff(v, half));
                    b.u[v] = ld2h(W + (size_t)(kb * 16 + l16) * HDc + 32 * c + b_koff(v, half));
                }
                acc = wmma_f16(a, b, acc);
            }
#pragma unroll
            for (int g = 0; g < 8; ++g)
                Ms[kb * 256 + (g + 8 * half) * 16 + l16] = (_Float16)acc[g];
        }
        __syncthreads();

        // phase 2: per column block J: GEMM update + diagonal recurrence
        for (int J = wv; J <= I; J += 16) {
            v8f acc = zf();
            if (J < I) {
                int nt = I - J, c = 0;
                for (; c + 1 < nt; c += 2) {
                    FragH a, b;
#pragma unroll
                    for (int v = 0; v < 8; ++v) {
                        a.u[v] = *reinterpret_cast<const unsigned*>(
                            &Ms[(J + c + (v >> 2)) * 256 + l16 * 16 + 8 * half + ((v & 3) << 1)]);
                        b.u[v] = ld2h(Ct + (size_t)(J * 16 + l16) * Tc + J * 16 + c * 16 + b_koff(v, half));
                    }
                    acc = wmma_f16(a, b, acc);
                }
                if (nt & 1) {
                    FragH a, b;
#pragma unroll
                    for (int v = 0; v < 8; ++v) {
                        a.u[v] = (v < 4) ? *reinterpret_cast<const unsigned*>(
                            &Ms[(J + c) * 256 + l16 * 16 + 8 * half + ((v & 3) << 1)]) : 0u;
                        b.u[v] = ld2h(Ct + (size_t)(J * 16 + l16) * Tc + J * 16 + c * 16 + b_koff(v, half));
                    }
                    acc = wmma_f16(a, b, acc);
                }
            }
            // gather full columns: lanes 0..15 own column j; rows 8..15 from lane+16
            float colv[16];
#pragma unroll
            for (int g = 0; g < 8; ++g) {
                float other = __shfl(acc[g], l16 + 16);
                colv[g] = acc[g];
                colv[8 + g] = other;
            }
            const int j = l16;
            float out[16];
#pragma unroll
            for (int r = 0; r < 16; ++r) {
                float rhs = (J < I) ? (float)Ms[J * 256 + r * 16 + j]
                                    : ((j < r) ? (float)Ms[I * 256 + r * 16 + j] : 0.f);
                float s = rhs - colv[r];
#pragma unroll
                for (int q = 0; q < r; ++q)
                    s -= (float)Ms[I * 256 + r * 16 + q] * out[q];
                s *= betas[I * 16 + r];
                if (J == I && j >= r) s = 0.f;
                out[r] = s;
            }
            if (lane < 16) {
                _Float16* dst = Ct + (size_t)(J * 16 + j) * Tc + I * 16;  // transposed, contiguous
#pragma unroll
                for (int r = 0; r < 16; ++r) dst[r] = (_Float16)out[r];
            }
        }
        __threadfence();
        __syncthreads();
    }
}

// ---------------------------------------------------------------- flash-style attention
// S = (tril(QK^T) - tril(QW^T) @ C') * HD^-0.5 ; online softmax ; O = A @ V
__global__ __launch_bounds__(128) void attn_k(
    const _Float16* __restrict__ qp, const _Float16* __restrict__ kp,
    const _Float16* __restrict__ vt, const _Float16* __restrict__ wp,
    const _Float16* __restrict__ Cws, _Float16* __restrict__ Op)
{
    constexpr int GP = 1032;                 // padded pitch (bank-conflict relief)
    __shared__ _Float16 Gs[4][16 * GP];      // per-wave tril(QW^T) strip
    __shared__ _Float16 Ps[4][16 * 32];      // per-wave probs staging (upper K half = 0)
    const int bh = blockIdx.x;
    const int lane = threadIdx.x & 31, wv = threadIdx.x >> 5;
    const int half = lane >> 4, l16 = lane & 15;
    const int Iblk = blockIdx.y * 4 + wv, I0 = Iblk * 16;
    const _Float16* qh = qp + (size_t)bh * Tc * HDc;
    const _Float16* kh = kp + (size_t)bh * Tc * HDc;
    const _Float16* vh = vt + (size_t)bh * HDc * Tc;   // [HD][T]
    const _Float16* wh = wp + (size_t)bh * Tc * HDc;
    const _Float16* Ct = Cws + (size_t)bh * Tc * Tc;   // transposed C'

    for (int e = lane; e < 256; e += 32)
        Ps[wv][(e >> 4) * 32 + 16 + (e & 15)] = (_Float16)0.f;

    FragH qf[2];
#pragma unroll
    for (int c = 0; c < 2; ++c)
#pragma unroll
        for (int v = 0; v < 8; ++v)
            qf[c].u[v] = ld2h(qh + (size_t)(I0 + l16) * HDc + 32 * c + a_koff(v, half));

    // G strip: QW^T masked t<=i
    for (int kt = 0; kt <= Iblk; ++kt) {
        v8f acc = zf();
#pragma unroll
        for (int c = 0; c < 2; ++c) {
            FragH b;
#pragma unroll
            for (int v = 0; v < 8; ++v)
                b.u[v] = ld2h(wh + (size_t)(kt * 16 + l16) * HDc + 32 * c + b_koff(v, half));
            acc = wmma_f16(qf[c], b, acc);
        }
#pragma unroll
        for (int g = 0; g < 8; ++g) {
            float val = acc[g];
            if (kt == Iblk) {
                int tcol = kt * 16 + l16, irow = I0 + g + 8 * half;
                if (tcol > irow) val = 0.f;
            }
            Gs[wv][(g + 8 * half) * GP + kt * 16 + l16] = (_Float16)val;
        }
    }

    float mrow[8], lrow[8];
#pragma unroll
    for (int g = 0; g < 8; ++g) { mrow[g] = -1e30f; lrow[g] = 0.f; }
    v8f Oacc[4];
#pragma unroll
    for (int dt = 0; dt < 4; ++dt) Oacc[dt] = zf();

    for (int J = 0; J <= Iblk; ++J) {
        const int t0 = J * 16;
        v8f sK = zf();
#pragma unroll
        for (int c = 0; c < 2; ++c) {
            FragH b;
#pragma unroll
            for (int v = 0; v < 8; ++v)
                b.u[v] = ld2h(kh + (size_t)(t0 + l16) * HDc + 32 * c + b_koff(v, half));
            sK = wmma_f16(qf[c], b, sK);
        }
        v8f sG = zf();
        int nt = Iblk + 1 - J, c = 0;
        for (; c + 1 < nt; c += 2) {
            FragH a, b;
#pragma unroll
            for (int v = 0; v < 8; ++v) {
                a.u[v] = *reinterpret_cast<const unsigned*>(&Gs[wv][l16 * GP + t0 + c * 16 + a_koff(v, half)]);
                b.u[v] = ld2h(Ct + (size_t)(t0 + l16) * Tc + t0 + c * 16 + b_koff(v, half));
            }
            sG = wmma_f16(a, b, sG);
        }
        if (nt & 1) {
            FragH a, b;
#pragma unroll
            for (int v = 0; v < 8; ++v) {
                a.u[v] = (v < 4) ? *reinterpret_cast<const unsigned*>(
                    &Gs[wv][l16 * GP + t0 + c * 16 + a_koff(v, half)]) : 0u;
                int kk = t0 + c * 16 + b_koff(v, half);
                if (kk > Tc - 2) kk = Tc - 2;   // x0 anyway (A upper half zero)
                b.u[v] = ld2h(Ct + (size_t)(t0 + l16) * Tc + kk);
            }
            sG = wmma_f16(a, b, sG);
        }
        // score + online softmax (rows live across 16-lane groups)
        float sv[8], tm[8];
#pragma unroll
        for (int g = 0; g < 8; ++g) {
            float s = (sK[g] - sG[g]) * 0.125f;
            if (J == Iblk) {
                int jcol = t0 + l16, irow = I0 + g + 8 * half;
                if (jcol > irow) s = -1e30f;
            }
            sv[g] = s; tm[g] = s;
        }
#pragma unroll
        for (int g = 0; g < 8; ++g)
            for (int m2 = 1; m2 < 16; m2 <<= 1) tm[g] = fmaxf(tm[g], __shfl_xor(tm[g], m2));
        float pr[8], rs[8];
#pragma unroll
        for (int g = 0; g < 8; ++g) {
            float mn = fmaxf(mrow[g], tm[g]);
            float corr = __expf(mrow[g] - mn);
            pr[g] = __expf(sv[g] - mn);
            rs[g] = pr[g];
            mrow[g] = mn;
            lrow[g] *= corr;
#pragma unroll
            for (int dt = 0; dt < 4; ++dt) Oacc[dt][g] *= corr;
        }
#pragma unroll
        for (int g = 0; g < 8; ++g) {
            for (int m2 = 1; m2 < 16; m2 <<= 1) rs[g] += __shfl_xor(rs[g], m2);
            lrow[g] += rs[g];
            Ps[wv][(g + 8 * half) * 32 + l16] = (_Float16)pr[g];
        }
        FragH pa;
#pragma unroll
        for (int v = 0; v < 8; ++v)
            pa.u[v] = *reinterpret_cast<const unsigned*>(&Ps[wv][l16 * 32 + a_koff(v, half)]);
#pragma unroll
        for (int dt = 0; dt < 4; ++dt) {
            FragH b;
#pragma unroll
            for (int v = 0; v < 8; ++v) {
                int kk = t0 + b_koff(v, half);
                if (kk > Tc - 2) kk = Tc - 2;   // K>=16 multiplied by zero probs
                b.u[v] = ld2h(vh + (size_t)(dt * 16 + l16) * Tc + kk);
            }
            Oacc[dt] = wmma_f16(pa, b, Oacc[dt]);
        }
    }
#pragma unroll
    for (int g = 0; g < 8; ++g) {
        float inv = 1.f / lrow[g];
#pragma unroll
        for (int dt = 0; dt < 4; ++dt)
            Op[(size_t)bh * Tc * HDc + (size_t)(I0 + g + 8 * half) * HDc + dt * 16 + l16] =
                (_Float16)(Oacc[dt][g] * inv);
    }
}

// ---------------------------------------------------------------- host orchestration
extern "C" void kernel_launch(void* const* d_in, const int* in_sizes, int n_in,
                              void* d_out, int out_size, void* d_ws, size_t ws_size,
                              hipStream_t stream)
{
    const float* x   = (const float*)d_in[0];
    const float* Wq  = (const float*)d_in[1];
    const float* Wk  = (const float*)d_in[2];
    const float* Wv  = (const float*)d_in[3];
    const float* Wwa = (const float*)d_in[4];
    const float* Wwb = (const float*)d_in[5];
    const float* cw  = (const float*)d_in[6];
    const float* Wb  = (const float*)d_in[7];
    const float* Wo  = (const float*)d_in[8];
    float* outp = (float*)d_out;

    char* base = (char*)d_ws;
    size_t off = 0;
    auto alloc = [&](size_t bytes) -> char* {
        char* p = base + off;
        off = (off + bytes + 255) & ~(size_t)255;
        return p;
    };
    _Float16* xh   = (_Float16*)alloc((size_t)BTc * Dc * 2);
    _Float16* WqT  = (_Float16*)alloc((size_t)Dc * Dc * 2);
    _Float16* WkT  = (_Float16*)alloc((size_t)Dc * Dc * 2);
    _Float16* WvT  = (_Float16*)alloc((size_t)Dc * Dc * 2);
    _Float16* WoT  = (_Float16*)alloc((size_t)Dc * Dc * 2);
    _Float16* WwaT = (_Float16*)alloc((size_t)RKc * Dc * 2);
    _Float16* WwbT = (_Float16*)alloc((size_t)Dc * RKc * 2);
    _Float16* WbT  = (_Float16*)alloc((size_t)Hc * Dc * 2);
    _Float16* qh   = (_Float16*)alloc((size_t)BHc * Tc * HDc * 2);
    _Float16* kh   = (_Float16*)alloc((size_t)BHc * Tc * HDc * 2);
    _Float16* vtb  = (_Float16*)alloc((size_t)BHc * HDc * Tc * 2);
    _Float16* th   = (_Float16*)alloc((size_t)BTc * RKc * 2);
    _Float16* w0h  = (_Float16*)alloc((size_t)BTc * Dc * 2);
    _Float16* wpb  = (_Float16*)alloc((size_t)BHc * Tc * HDc * 2);
    _Float16* betal= (_Float16*)alloc((size_t)BTc * Hc * 2);
    float*    betaf= (float*)   alloc((size_t)BHc * Tc * 4);
    _Float16* Cws  = (_Float16*)alloc((size_t)BHc * Tc * Tc * 2 + 4096);
    _Float16* Oph  = (_Float16*)alloc((size_t)BHc * Tc * HDc * 2);
    (void)ws_size; (void)in_sizes; (void)n_in; (void)out_size;

    // casts / transposes to f16
    cast_f16_k<<<2048, 256, 0, stream>>>(x, xh, BTc * Dc);
    cast_tr_k<<<4096, 256, 0, stream>>>(Wq, WqT, Dc, Dc);
    cast_tr_k<<<4096, 256, 0, stream>>>(Wk, WkT, Dc, Dc);
    cast_tr_k<<<4096, 256, 0, stream>>>(Wv, WvT, Dc, Dc);
    cast_tr_k<<<4096, 256, 0, stream>>>(Wo, WoT, Dc, Dc);
    cast_tr_k<<<128, 256, 0, stream>>>(Wwa, WwaT, Dc, RKc);
    cast_tr_k<<<128, 256, 0, stream>>>(Wwb, WwbT, RKc, Dc);
    cast_tr_k<<<64, 256, 0, stream>>>(Wb, WbT, Dc, Hc);

    // projections (WMMA, LDS-staged B with async double-buffer)
    gemm_wmma2_k<0, 1><<<dim3(16, 16), 256, 0, stream>>>(xh, WqT, qh, nullptr, BTc, Dc, Dc, Dc, Dc);
    gemm_wmma2_k<0, 1><<<dim3(16, 16), 256, 0, stream>>>(xh, WkT, kh, nullptr, BTc, Dc, Dc, Dc, Dc);
    gemm_wmma2_k<0, 2><<<dim3(16, 16), 256, 0, stream>>>(xh, WvT, vtb, nullptr, BTc, Dc, Dc, Dc, Dc);
    gemm_small_k<<<dim3(64, 1), 256, 0, stream>>>(xh, WwaT, th, BTc, RKc, Dc, Dc, Dc);
    gemm_wmma2_k<0, 0><<<dim3(16, 16), 256, 0, stream>>>(th, WwbT, w0h, nullptr, BTc, Dc, RKc, RKc, RKc);
    gemm_small_k<<<dim3(64, 1), 256, 0, stream>>>(xh, WbT, betal, BTc, Hc, Dc, Dc, Dc);

    // elementwise: conv+silu+norm, beta
    conv_silu_norm_k<<<4096, 256, 0, stream>>>(w0h, cw, wpb);
    beta_k<<<128, 256, 0, stream>>>(betal, betaf);

    // triangular solve for C' (per head)
    solve_k<<<32, 512, 0, stream>>>(wpb, betaf, Cws);

    // attention core
    attn_k<<<dim3(32, 16), 128, 0, stream>>>(qh, kh, vtb, wpb, Cws, Oph);

    // output projection -> f32 d_out
    gemm_wmma2_k<1, 3><<<dim3(16, 16), 256, 0, stream>>>(Oph, WoT, nullptr, outp, BTc, Dc, Dc, 0, Dc);
}
